// TokenClusteringBlock_1752346657614
// MI455X (gfx1250) — compile-verified, matrix operand order checked by reading, not run on hardware
//
#include <hip/hip_runtime.h>

// ---------------------------------------------------------------------------
// Token clustering (superpixel SLIC-like) for MI455X / gfx1250.
// B=8, H=W=96, C=384, 24x24 superpixel grid, 5x5 window, 5 iterations.
// f16 WMMA (v_wmma_f32_16x16x32_f16) for all dot products; fp32 softmax,
// fp32 atomic accumulation, fp32 centroids. Problem is L2-resident (113MB
// input < 192MB L2), mask sparsity (25/576) exploited by tiling per-cell.
// Candidate-centroid staging uses GLOBAL_LOAD_ASYNC_TO_LDS_B128 when the
// toolchain exposes it (ASYNCcnt path), else a plain vector copy.
// ---------------------------------------------------------------------------

#define B_    8
#define H_    96
#define W_    96
#define C_    384
#define SG    24
#define M_    (SG * SG)     // 576 superpixels
#define NPIX  (H_ * W_)     // 9216 pixels
#define TEMP_ 0.01f
#define NITER 5

typedef __attribute__((ext_vector_type(16))) _Float16 v16h;
typedef __attribute__((ext_vector_type(8)))  float    v8f;
typedef __attribute__((ext_vector_type(4)))  int      v4i;

// --- gfx1250 async global->LDS copy (bytes; both buffers already f16) ------
#if defined(__has_builtin)
#if __has_builtin(__builtin_amdgcn_global_load_async_to_lds_b128)
#define HAVE_ASYNC_LDS 1
#endif
#endif

typedef __attribute__((address_space(1))) v4i gv4i;   // global int4
typedef __attribute__((address_space(3))) v4i lv4i;   // LDS int4

// Global flat address == AS1 address (identity map).
__device__ __forceinline__ gv4i* as_global_v4(const void* p) {
  return (gv4i*)(unsigned long long)p;
}
// Flat LDS address: addr[31:0] is the DS byte offset (ISA 10.2 aperture map).
__device__ __forceinline__ lv4i* as_local_v4(const void* p) {
  return (lv4i*)(unsigned int)(unsigned long long)p;
}

__device__ __forceinline__ void stage16(const void* gsrc, void* ldst) {
#if HAVE_ASYNC_LDS
  // 16B per lane, global -> LDS, no VGPR round trip; tracked on ASYNCcnt.
  __builtin_amdgcn_global_load_async_to_lds_b128(as_global_v4(gsrc),
                                                 as_local_v4(ldst), 0, 0);
#else
  *(uint4*)ldst = *(const uint4*)gsrc;
#endif
}

__device__ __forceinline__ void stage_wait() {
#if HAVE_ASYNC_LDS
#if __has_builtin(__builtin_amdgcn_s_wait_asynccnt)
  __builtin_amdgcn_s_wait_asynccnt(0);
#else
  asm volatile("s_wait_asynccnt 0x0" ::: "memory");
#endif
#endif
}

// A/B operand fragment for V_WMMA_F32_16X16X32_F16 from a row-major
// [row][stride] half buffer (ISA 7.12.2 16-bit A-matrix 16x32 layout):
//   lanes 0-15 : row = lane,    elems 0-7 = K kb+0..7,  elems 8-15 = K kb+16..23
//   lanes16-31 : row = lane-16, elems 0-7 = K kb+8..15, elems 8-15 = K kb+24..31
__device__ __forceinline__ v16h frag_ld_row(const _Float16* base, int stride,
                                            int row, int kb, int lane) {
  const int hi = (lane >> 4) & 1;
  const _Float16* p = base + (size_t)row * stride + kb + hi * 8;
  v16h v;
#pragma unroll
  for (int i = 0; i < 8; ++i) v[i] = p[i];
#pragma unroll
  for (int i = 0; i < 8; ++i) v[8 + i] = p[16 + i];
  return v;
}

// Same fragment pattern but operand stored [k][col] (column read, kb=0).
__device__ __forceinline__ v16h frag_ld_col(const _Float16* base, int stride,
                                            int col, int lane) {
  const int hi = (lane >> 4) & 1;
  const _Float16* p = base + col + (size_t)(hi * 8) * stride;
  v16h v;
#pragma unroll
  for (int i = 0; i < 8; ++i) v[i] = p[(size_t)i * stride];
#pragma unroll
  for (int i = 0; i < 8; ++i) v[8 + i] = p[(size_t)(16 + i) * stride];
  return v;
}

// ---------------------------------------------------------------------------
// K0: init centroids = 4x4 block means; write fp32 sp (into d_out), f16 copy,
// and squared norms. One block per (b, m), 128 threads over channels.
// ---------------------------------------------------------------------------
__global__ __launch_bounds__(128) void k_init(const float* __restrict__ pf,
                                              float* __restrict__ sp,
                                              _Float16* __restrict__ sph,
                                              float* __restrict__ snorm) {
  const int b = blockIdx.x / M_, m = blockIdx.x % M_;
  const int sh = m / SG, sw = m % SG;
  const int t = threadIdx.x;
  __shared__ float red[128];
  float ssq = 0.f;
  for (int c = t; c < C_; c += 128) {
    float s = 0.f;
#pragma unroll
    for (int i = 0; i < 16; ++i) {
      const int y = sh * 4 + (i >> 2), x = sw * 4 + (i & 3);
      s += pf[(((size_t)b * H_ + y) * W_ + x) * C_ + c];
    }
    s *= (1.f / 16.f);
    const size_t o = ((size_t)b * M_ + m) * C_ + c;
    sp[o] = s;
    sph[o] = (_Float16)s;
    ssq += s * s;
  }
  red[t] = ssq;
  __syncthreads();
  for (int o = 64; o > 0; o >>= 1) {
    if (t < o) red[t] += red[t + o];
    __syncthreads();
  }
  if (t == 0) snorm[b * M_ + m] = red[0];
}

// ---------------------------------------------------------------------------
// Zero accum + denom (contiguous in ws).
// ---------------------------------------------------------------------------
__global__ void k_zero(float* __restrict__ p, size_t n) {
  const size_t i = (size_t)blockIdx.x * blockDim.x + threadIdx.x;
  if (i < n) p[i] = 0.f;
}

// ---------------------------------------------------------------------------
// K2: one workgroup per (b, 4x4 cell). GEMM1 (4-wave split-K WMMA):
// dot(16 pix x 32 cand); softmax over candidates; GEMM2 (WMMA):
// aff^T x pix -> centroid contributions, scattered with fp32 atomics.
// ---------------------------------------------------------------------------
__global__ __launch_bounds__(128) void k_iter(const float* __restrict__ pf,
                                              const _Float16* __restrict__ sph,
                                              const float* __restrict__ snorm,
                                              float* __restrict__ accum,
                                              float* __restrict__ denom) {
  __shared__ alignas(16) _Float16 pixh[32][C_]; // rows 16..31 zero (K pad, GEMM2)
  __shared__ alignas(16) _Float16 spc[32][C_];  // 25 candidates padded to 32
  __shared__ float    sc[16][32];               // K-half-0 dots, then affinities
  __shared__ float    sc2[16][32];              // K-half-1 dots
  __shared__ alignas(16) _Float16 affT[32][32]; // aff^T, K(pixel) padded to 32
  __shared__ float    snL[32];
  __shared__ int      cid[32];

  const int b = blockIdx.x / M_;
  const int cell = blockIdx.x % M_;
  const int cy = cell / SG, cx = cell % SG;
  const int t = threadIdx.x;
  const int wave = t >> 5, lane = t & 31;

  if (t < 32) {
    int id = -1;
    float sn = 0.f;
    if (t < 25) {
      const int yy = cy + t / 5 - 2, xx = cx + t % 5 - 2;
      if (yy >= 0 && yy < SG && xx >= 0 && xx < SG) {
        id = yy * SG + xx;
        sn = snorm[b * M_ + id];
      }
    }
    cid[t] = id;
    snL[t] = sn;
  }
  __syncthreads();

  // Stage candidate centroids: 16B chunks, async global->LDS when available.
  // 32 rows x 48 chunks (384 halfs / 8) = 1536 chunks, 12 per thread.
  for (int ch = t; ch < 32 * 48; ch += 128) {
    const int j = ch / 48, c = (ch % 48) * 8;
    const int id = cid[j];
    if (id >= 0) {
      stage16(sph + ((size_t)b * M_ + id) * C_ + c, &spc[j][c]);
    } else {
      *(uint4*)&spc[j][c] = uint4{0u, 0u, 0u, 0u};
    }
  }
  // Stage pixel tile (f32 -> f16 conversion, overlaps with async copies).
  for (int idx = t; idx < 32 * C_; idx += 128) {
    const int r = idx / C_, c = idx % C_;
    float v = 0.f;
    if (r < 16) {
      const int y = cy * 4 + (r >> 2), x = cx * 4 + (r & 3);
      v = pf[(((size_t)b * H_ + y) * W_ + x) * C_ + c];
    }
    pixh[r][c] = (_Float16)v;
  }
  stage_wait();
  __syncthreads();

  // --- GEMM1: dot[pix 16][cand 32], K=384. All 4 waves: wave = (Khalf, tile),
  // 6 WMMA steps each; partials in sc (Khalf 0) / sc2 (Khalf 1).
  {
    const int ctile = wave & 1;
    const int kh = wave >> 1;
    v8f acc = {};
    for (int ks = kh * 6; ks < kh * 6 + 6; ++ks) {
      const v16h a  = frag_ld_row(&pixh[0][0], C_, lane & 15, ks * 32, lane);
      const v16h bf = frag_ld_row(&spc[0][0], C_, ctile * 16 + (lane & 15),
                                  ks * 32, lane);
      acc = __builtin_amdgcn_wmma_f32_16x16x32_f16(false, a, false, bf,
                                                   (short)0, acc, false, false);
    }
    float* dst = kh ? &sc2[0][0] : &sc[0][0];
    const int prow = (lane >> 4) ? 8 : 0;  // C/D: VGPR r -> M=r / r+8
#pragma unroll
    for (int r = 0; r < 8; ++r)
      dst[(prow + r) * 32 + ctile * 16 + (lane & 15)] = acc[r];
  }
  __syncthreads();

  // --- Softmax over candidates (||pix||^2 cancels; logit = (2*dot-||sp||^2)*T).
  if (t < 16) {
    float lg[32];
    float mx = -3.0e38f;
#pragma unroll
    for (int j = 0; j < 32; ++j) {
      const float dot = sc[t][j] + sc2[t][j];
      const float l = (cid[j] >= 0) ? (2.f * dot - snL[j]) * TEMP_ : -3.0e38f;
      lg[j] = l;
      mx = fmaxf(mx, l);
    }
    float s = 0.f;
#pragma unroll
    for (int j = 0; j < 32; ++j) {
      const float e = (cid[j] >= 0) ? __expf(lg[j] - mx) : 0.f;
      lg[j] = e;
      s += e;
    }
    const float inv = 1.f / fmaxf(s, 1e-30f);
#pragma unroll
    for (int j = 0; j < 32; ++j) {
      const float a = lg[j] * inv;
      sc[t][j] = a;                       // affinity (fp32)
      affT[j][t] = (_Float16)a;           // transposed, f16 for GEMM2
      affT[j][t + 16] = (_Float16)0.f;    // K padding (pixels 16..31)
    }
  }
  __syncthreads();

  // --- Denominator partial sums.
  if (t < 32 && cid[t] >= 0) {
    float s = 0.f;
#pragma unroll
    for (int p = 0; p < 16; ++p) s += sc[p][t];
    atomicAdd(&denom[b * M_ + cid[t]], s);
  }

  // --- GEMM2: contrib[cand 32][C 384] = affT(32x32, K=pix) x pixh(32x384).
  for (int ct = wave; ct < C_ / 16; ct += 4) {
    const v16h b2 = frag_ld_col(&pixh[0][0], C_, ct * 16 + (lane & 15), lane);
#pragma unroll
    for (int mt = 0; mt < 2; ++mt) {
      const v16h a2 = frag_ld_row(&affT[0][0], 32, mt * 16 + (lane & 15), 0, lane);
      v8f acc = {};
      acc = __builtin_amdgcn_wmma_f32_16x16x32_f16(false, a2, false, b2,
                                                   (short)0, acc, false, false);
      const int rbase = mt * 16 + ((lane >> 4) ? 8 : 0);
      const int ccol = ct * 16 + (lane & 15);
#pragma unroll
      for (int r = 0; r < 8; ++r) {
        const int id = cid[rbase + r];
        if (id >= 0)
          atomicAdd(&accum[((size_t)b * M_ + id) * C_ + ccol], acc[r]);
      }
    }
  }
}

// ---------------------------------------------------------------------------
// K3: sp = accum / max(denom, 1e-16); refresh f16 copy + norms.
// ---------------------------------------------------------------------------
__global__ __launch_bounds__(128) void k_finalize(const float* __restrict__ accum,
                                                  const float* __restrict__ denom,
                                                  float* __restrict__ sp,
                                                  _Float16* __restrict__ sph,
                                                  float* __restrict__ snorm) {
  const int b = blockIdx.x / M_, m = blockIdx.x % M_;
  const int t = threadIdx.x;
  __shared__ float red[128];
  const float inv = 1.f / fmaxf(denom[b * M_ + m], 1e-16f);
  float ssq = 0.f;
  for (int c = t; c < C_; c += 128) {
    const size_t o = ((size_t)b * M_ + m) * C_ + c;
    const float s = accum[o] * inv;
    sp[o] = s;
    sph[o] = (_Float16)s;
    ssq += s * s;
  }
  red[t] = ssq;
  __syncthreads();
  for (int o = 64; o > 0; o >>= 1) {
    if (t < o) red[t] += red[t + o];
    __syncthreads();
  }
  if (t == 0) snorm[b * M_ + m] = red[0];
}

// ---------------------------------------------------------------------------
// K4: hard labels = argmin over all 576 superpixels (unmasked final dist).
// One workgroup per 16-pixel tile; dense WMMA over 36 sp tiles, K=384.
// ---------------------------------------------------------------------------
__global__ __launch_bounds__(128) void k_labels(const float* __restrict__ pf,
                                                const _Float16* __restrict__ sph,
                                                const float* __restrict__ snorm,
                                                float* __restrict__ olab) {
  __shared__ alignas(16) _Float16 pixh[16][C_];
  __shared__ float scores[16][M_];

  const int b = blockIdx.x / (NPIX / 16);
  const int pt = blockIdx.x % (NPIX / 16);
  const int t = threadIdx.x, wave = t >> 5, lane = t & 31;

  for (int idx = t; idx < 16 * C_; idx += 128) {
    const int r = idx / C_, c = idx % C_;
    pixh[r][c] = (_Float16)pf[((size_t)b * NPIX + pt * 16 + r) * C_ + c];
  }
  __syncthreads();

  for (int st = wave; st < M_ / 16; st += 4) {
    const int m = st * 16 + (lane & 15);
    const _Float16* srow = sph + ((size_t)b * M_ + m) * C_;
    if (st + 4 < M_ / 16)
      __builtin_prefetch(sph + ((size_t)b * M_ + (st + 4) * 16 + (lane & 15)) * C_,
                         0, 1);
    v8f acc = {};
    for (int ks = 0; ks < 12; ++ks) {
      const v16h a  = frag_ld_row(&pixh[0][0], C_, lane & 15, ks * 32, lane);
      const v16h bf = frag_ld_row(srow, 0, 0, ks * 32, lane);  // global B frag
      acc = __builtin_amdgcn_wmma_f32_16x16x32_f16(false, a, false, bf,
                                                   (short)0, acc, false, false);
    }
    const float sn = snorm[b * M_ + m];
    const int prow = (lane >> 4) ? 8 : 0;
#pragma unroll
    for (int r = 0; r < 8; ++r)
      scores[prow + r][m] = sn - 2.f * acc[r];  // ||pix||^2 constant in argmin
  }
  __syncthreads();

  if (t < 16) {
    float best = 3.4e38f;
    int bi = 0;
    for (int m = 0; m < M_; ++m) {
      const float s = scores[t][m];
      if (s < best) { best = s; bi = m; }  // strict < keeps first-min (jnp.argmin)
    }
    olab[(size_t)b * NPIX + pt * 16 + t] = (float)bi;
  }
}

// ---------------------------------------------------------------------------
extern "C" void kernel_launch(void* const* d_in, const int* in_sizes, int n_in,
                              void* d_out, int out_size, void* d_ws, size_t ws_size,
                              hipStream_t stream) {
  (void)in_sizes; (void)n_in; (void)out_size; (void)ws_size;
  const float* pf = (const float*)d_in[0];

  // d_out: [spixel_features (B,24,24,C) fp32][hard_labels (B,N) as float]
  float* sp = (float*)d_out;                       // centroids live here
  float* olab = sp + (size_t)B_ * M_ * C_;

  // workspace: [accum BMC][denom BM][snorm BM][sph BMC halfs] ~= 10.2 MB
  float* accum = (float*)d_ws;
  float* denom = accum + (size_t)B_ * M_ * C_;
  float* snorm = denom + (size_t)B_ * M_;
  _Float16* sph = (_Float16*)(snorm + (size_t)B_ * M_);

  k_init<<<B_ * M_, 128, 0, stream>>>(pf, sp, sph, snorm);

  const size_t nz = (size_t)B_ * M_ * C_ + (size_t)B_ * M_;  // accum + denom
  const int zblocks = (int)((nz + 255) / 256);
  for (int it = 0; it < NITER; ++it) {
    k_zero<<<zblocks, 256, 0, stream>>>(accum, nz);
    k_iter<<<B_ * M_, 128, 0, stream>>>(pf, sph, snorm, accum, denom);
    k_finalize<<<B_ * M_, 128, 0, stream>>>(accum, denom, sp, sph, snorm);
  }

  k_labels<<<B_ * (NPIX / 16), 128, 0, stream>>>(pf, sph, snorm, olab);
}
